// Neighbor_PairConLoss_68152541053488
// MI455X (gfx1250) — matrix operand order repaired and checked by reading, not compiled, not run on hardware
//
#include <hip/hip_runtime.h>

typedef __bf16 bf16;
typedef __bf16 v8bf  __attribute__((ext_vector_type(8)));
typedef __bf16 v16bf __attribute__((ext_vector_type(16)));
typedef float  v8f   __attribute__((ext_vector_type(8)));

#define NROWS 8192     // 2*M
#define DK    256      // feature dim
#define M_IDX 4096     // M
#define INV_T 20.0f    // 1/0.05
#define CTILE 32       // columns staged into LDS per step
#define COLS_PER_BLOCK 1024

// ---------------------------------------------------------------------------
// CDNA5 async-copy helpers (gfx1250). Inline asm keeps this portable across
// ROCm-7.2 and amdgpu-toolchain. LDS destination = low 32 bits of the generic
// shared pointer (ISA 10.2: LDS aperture addr[31:0] is the LDS offset).
// ---------------------------------------------------------------------------
__device__ __forceinline__ void async_load_b128(uint32_t lds_off, const void* gptr) {
    asm volatile("global_load_async_to_lds_b128 %0, %1, off"
                 :: "v"(lds_off), "v"(gptr)
                 : "memory");
}
__device__ __forceinline__ void wait_asynccnt0() {
#if __has_builtin(__builtin_amdgcn_s_wait_asynccnt)
    __builtin_amdgcn_s_wait_asynccnt(0);
#else
    asm volatile("s_wait_asynccnt 0x0" ::: "memory");
#endif
}

// ---------------------------------------------------------------------------
// Kernel 1: gather rows by index, cast fp32 -> bf16, zero accumulators.
// ---------------------------------------------------------------------------
__global__ __launch_bounds__(256) void gather_cast_kernel(
    const float* __restrict__ feat1, const float* __restrict__ feat2,
    const int* __restrict__ idx1, const int* __restrict__ idx2,
    bf16* __restrict__ pos, bf16* __restrict__ neg,
    float* __restrict__ Ng, float* __restrict__ Pg)
{
    const int r = blockIdx.x;
    const int t = threadIdx.x;
    const bool second = (r >= M_IDX);
    const int m = second ? (r - M_IDX) : r;
    const float* base = second ? feat2 : feat1;

    const size_t poff = (size_t)idx1[m] * DK;
    const size_t noff = (size_t)idx2[m] * DK;
    pos[(size_t)r * DK + t] = (bf16)base[poff + t];
    neg[(size_t)r * DK + t] = (bf16)base[noff + t];
    if (t == 0) { Ng[r] = 0.0f; Pg[r] = 0.0f; }
}

// ---------------------------------------------------------------------------
// Kernel 2: rowsum[i] += sum_j exp( (A_i . B_j) * INV_T ) over a column slice.
//
// Block = 256 threads = 8 wave32s; wave w owns rows [blockIdx.x*128 + w*16).
// A stripe (16x256 bf16) register-resident per wave.
// B: 32-column tiles (32 contiguous rows of B = linear 16 KiB) staged into a
// double-buffered LDS tile with global_load_async_to_lds_b128, consumed by all
// 8 waves via ds_load_b128 — removes the 8x redundant L2 fetch of B.
// ---------------------------------------------------------------------------
__global__ __launch_bounds__(256) void sim_exp_rowsum_kernel(
    const bf16* __restrict__ A, const bf16* __restrict__ B,
    float* __restrict__ rowsum)
{
    __shared__ __align__(128) bf16 Bsh[2][CTILE * DK];   // 2 x 16 KiB

    const int tid  = threadIdx.x;
    const int lane = tid & 31;
    const int wave = tid >> 5;
    const int row0 = blockIdx.x * 128 + wave * 16;
    const int col_begin = blockIdx.y * COLS_PER_BLOCK;

    // A fragment layout (16-bit A 16x32, wave32):
    //   lanes 0-15 : M=lane,    K = kb+0..7  and kb+16..23
    //   lanes16-31 : M=lane-16, K = kb+8..15 and kb+24..31
    const int aRow = row0 + (lane & 15);
    const int aSel = (lane < 16) ? 0 : 8;
    // B fragment layout (16-bit B 32x16, wave32):
    //   lanes 0-15 : N=lane,    K = kb+0..7 and kb+16..23 ; lanes16-31 : N=lane-16, K=kb+8..15 and kb+24..31
    const int bSel = (lane < 16) ? 0 : 8;

    // Preload this wave's A stripe: 8 fragments covering K=0..255.
    const bf16* arow = A + (size_t)aRow * DK;
    v16bf afrag[8];
#pragma unroll
    for (int kk = 0; kk < 8; ++kk) {
        const int kb = kk * 32;
        v8bf lo = *(const v8bf*)(arow + kb + aSel);
        v8bf hi = *(const v8bf*)(arow + kb + 16 + aSel);
        afrag[kk] = __builtin_shufflevector(
            lo, hi, 0, 1, 2, 3, 4, 5, 6, 7, 8, 9, 10, 11, 12, 13, 14, 15);
    }

    float acc[8];
#pragma unroll
    for (int v = 0; v < 8; ++v) acc[v] = 0.0f;

    // Stage one CTILE of B (16 KiB linear): 256 threads x 4 async b128 each.
    auto stage = [&](int buf, int c0) {
        const char* gsrc = (const char*)(B + (size_t)c0 * DK);
        const uint32_t lbase = (uint32_t)(uintptr_t)(&Bsh[buf][0]);
#pragma unroll
        for (int i = 0; i < 4; ++i) {
            const uint32_t off = (uint32_t)(tid + i * 256) * 16u;
            async_load_b128(lbase + off, gsrc + off);
        }
    };

    stage(0, col_begin);
    wait_asynccnt0();
    __syncthreads();

    for (int it = 0; it < COLS_PER_BLOCK / CTILE; ++it) {
        const int cur = it & 1;
        const int c0  = col_begin + it * CTILE;
        if (it + 1 < COLS_PER_BLOCK / CTILE)
            stage(cur ^ 1, c0 + CTILE);          // prefetch next tile -> LDS

        // Two 16-column subtiles from the staged LDS tile.
#pragma unroll
        for (int sub = 0; sub < 2; ++sub) {
            const bf16* bl = &Bsh[cur][(sub * 16 + (lane & 15)) * DK + bSel];
            v8f cacc = {};
#pragma unroll
            for (int kk = 0; kk < 8; ++kk) {
                v8bf b0 = *(const v8bf*)(bl + kk * 32);      // ds_load_b128
                v8bf b1 = *(const v8bf*)(bl + kk * 32 + 16); // ds_load_b128
                v16bf bfragv = __builtin_shufflevector(
                    b0, b1, 0, 1, 2, 3, 4, 5, 6, 7, 8, 9, 10, 11, 12, 13, 14, 15);
                cacc = __builtin_amdgcn_wmma_f32_16x16x32_bf16(
                    false, afrag[kk], false, bfragv, (short)0, cacc, false, false);
            }
            // C layout: VGPR v -> row row0+v (lanes 0-15) / row0+v+8 (16-31);
            // column = c0 + sub*16 + (lane&15).
#pragma unroll
            for (int v = 0; v < 8; ++v)
                acc[v] += __expf(cacc[v] * INV_T);  // v_exp_f32 co-executes w/ WMMA
        }

        wait_asynccnt0();    // my async loads for the next buffer are done
        __syncthreads();     // everyone finished reading cur + loading next
    }

    // Reduce the 16 column-lanes within each half-wave, one atomic per row.
#pragma unroll
    for (int v = 0; v < 8; ++v) {
        float s = acc[v];
        s += __shfl_xor(s, 1, 16);
        s += __shfl_xor(s, 2, 16);
        s += __shfl_xor(s, 4, 16);
        s += __shfl_xor(s, 8, 16);
        if ((lane & 15) == 0) {
            const int row = row0 + v + ((lane >= 16) ? 8 : 0);
            atomicAdd(&rowsum[row], s);
        }
    }
}

// ---------------------------------------------------------------------------
// Kernel 3: loss = mean_i( -log( Pg_i / (Ng_i + Pg_i) ) )
// ---------------------------------------------------------------------------
__global__ __launch_bounds__(256) void loss_kernel(
    const float* __restrict__ Ng, const float* __restrict__ Pg,
    float* __restrict__ out)
{
    __shared__ float red[8];
    float s = 0.0f;
    for (int i = threadIdx.x; i < NROWS; i += 256) {
        const float pg = Pg[i];
        const float ng = Ng[i];
        s += -__logf(pg / (ng + pg));
    }
#pragma unroll
    for (int m = 1; m < 32; m <<= 1) s += __shfl_xor(s, m, 32);
    const int lane = threadIdx.x & 31;
    const int wave = threadIdx.x >> 5;
    if (lane == 0) red[wave] = s;
    __syncthreads();
    if (threadIdx.x == 0) {
        float t = 0.0f;
#pragma unroll
        for (int w = 0; w < 8; ++w) t += red[w];
        out[0] = t / (float)NROWS;
    }
}

// ---------------------------------------------------------------------------
// Launcher. Workspace: pos bf16 4MiB | neg bf16 4MiB | Ng f32 32KiB | Pg f32.
// ---------------------------------------------------------------------------
extern "C" void kernel_launch(void* const* d_in, const int* in_sizes, int n_in,
                              void* d_out, int out_size, void* d_ws, size_t ws_size,
                              hipStream_t stream)
{
    const float* feat1 = (const float*)d_in[0];
    const float* feat2 = (const float*)d_in[1];
    const int*   idx1  = (const int*)d_in[2];
    const int*   idx2  = (const int*)d_in[3];
    float* out = (float*)d_out;

    char* ws = (char*)d_ws;
    const size_t mat_bytes = (size_t)NROWS * DK * sizeof(bf16);
    bf16*  pos = (bf16*)ws;
    bf16*  neg = (bf16*)(ws + mat_bytes);
    float* Ng  = (float*)(ws + 2 * mat_bytes);
    float* Pg  = Ng + NROWS;

    gather_cast_kernel<<<NROWS, 256, 0, stream>>>(feat1, feat2, idx1, idx2,
                                                  pos, neg, Ng, Pg);

    dim3 grid(NROWS / 128, NROWS / COLS_PER_BLOCK);  // 64 x 8
    sim_exp_rowsum_kernel<<<grid, 256, 0, stream>>>(pos, neg, Ng);  // Ng
    sim_exp_rowsum_kernel<<<grid, 256, 0, stream>>>(pos, pos, Pg);  // Pg

    loss_kernel<<<1, 256, 0, stream>>>(Ng, Pg, out);
}
